// RNN_Part2_15109694948108
// MI455X (gfx1250) — matrix operand order repaired and checked by reading, not compiled, not run on hardware
//
#include <hip/hip_runtime.h>

// ---------------------------------------------------------------------------
// GRU seq2seq for MI455X (gfx1250).
// GEMMs run on v_wmma_f32_16x16x32_bf16 (wave32, 16x16 tiles, f32 accum).
// Per-wave: one 16x16 M-tile x 4 N-tiles (16x64 strip) sharing the A fragment.
// Weights are pre-packed into WMMA fragment-major bf16 layout so each lane's
// B fragment is one 32B contiguous load; A fragments are two b128 loads.
// ---------------------------------------------------------------------------

#define D_MODEL 800
#define BATCH   64
#define TSEQ    48
#define ROWS    (BATCH * TSEQ)   // 3072
#define H3      (3 * D_MODEL)    // 2400
#define VOUT    32000

typedef __attribute__((ext_vector_type(16))) __bf16 bf16x16;
typedef __attribute__((ext_vector_type(8)))  __bf16 bf16x8;
typedef __attribute__((ext_vector_type(8)))  float  f32x8;

__device__ __forceinline__ unsigned short f32_to_bf16(float f) {
    unsigned int u = __builtin_bit_cast(unsigned int, f);
    u += 0x7FFFu + ((u >> 16) & 1u);      // round-to-nearest-even
    return (unsigned short)(u >> 16);
}

// ---------------------------------------------------------------------------
// Pack fp32 B[K,N] -> bf16 fragment-major layout:
//   dst[((ntj*KT + kt)*32 + lane)*16 + e] = B[(kt*32 + (lane>>4)*16 + e)*N + ntj*16 + (lane&15)]
// so in the GEMM each lane loads its 16 B elements as 32 contiguous bytes,
// and consecutive K-chunks of one N-tile are contiguous (streaming).
// ---------------------------------------------------------------------------
__global__ void __launch_bounds__(256)
pack_B_bf16(const float* __restrict__ src, unsigned short* __restrict__ dst,
            int K, int N)
{
    const int KT = K >> 5;
    const long long total = (long long)K * N;
    long long i = (long long)blockIdx.x * 256 + threadIdx.x;
    const long long stride = (long long)gridDim.x * 256;
    for (; i < total; i += stride) {
        const int e    = (int)(i & 15);
        const int lane = (int)((i >> 4) & 31);
        const long long tile = i >> 9;
        const int kt  = (int)(tile % KT);
        const int ntj = (int)(tile / KT);
        const int k = (kt << 5) + ((lane >> 4) << 4) + e;
        const int n = (ntj << 4) + (lane & 15);
        dst[i] = f32_to_bf16(src[(long long)k * N + n]);
    }
}

// ---------------------------------------------------------------------------
// bf16 WMMA GEMM:  C[M,N] = A[M,K] * Bpacked + bias (optional relu)
// A row-major bf16; B pre-packed (see pack_B_bf16).
// M,N multiples of 16, ntN >= 4, K multiple of 32 (all call sites comply).
// Last wave per N-row overlaps instead of predicating: identical values are
// rewritten, keeping EXEC all-1s around every WMMA (ISA 7.12 requirement).
// ---------------------------------------------------------------------------
__global__ void __launch_bounds__(256)
gemm_bf16_wmma(const unsigned short* __restrict__ A,
               const unsigned short* __restrict__ Bp,
               const float* __restrict__ bias,
               float* __restrict__ Cf,
               unsigned short* __restrict__ Cb,
               int M, int N, int K, int relu)
{
    const int lane = threadIdx.x & 31;
    const int wid  = (int)(((unsigned)blockIdx.x * blockDim.x + threadIdx.x) >> 5);
    const int ntM = M >> 4;
    const int ntN = N >> 4;
    const int KT  = K >> 5;
    const int wavesN = (ntN + 3) >> 2;
    if (wid >= ntM * wavesN) return;           // wave-uniform
    const int tm = wid / wavesN;
    const int wn = wid - tm * wavesN;
    const int m0 = tm << 4;
    int nt0 = wn << 2;
    if (nt0 + 4 > ntN) nt0 = ntN - 4;          // overlap clamp (benign rewrite)

    const int half = lane >> 4;
    const int l15  = lane & 15;

    f32x8 acc0 = {}, acc1 = {}, acc2 = {}, acc3 = {};

    const unsigned short* arow = A + (size_t)(m0 + l15) * K + (half << 4);
    const size_t tileStride = (size_t)KT << 9;           // elems per N-tile stream
    const unsigned short* b0 = Bp + (size_t)nt0 * tileStride + ((size_t)lane << 4);
    const unsigned short* b1 = b0 + tileStride;
    const unsigned short* b2 = b1 + tileStride;
    const unsigned short* b3 = b2 + tileStride;

    for (int kt = 0; kt < KT; ++kt) {
        const unsigned short* ap = arow + (kt << 5);
        const bf16x8 alo = *(const bf16x8*)(ap);         // K = half*16 + 0..7
        const bf16x8 ahi = *(const bf16x8*)(ap + 8);     // K = half*16 + 8..15
        const bf16x16 a = __builtin_shufflevector(alo, ahi,
            0, 1, 2, 3, 4, 5, 6, 7, 8, 9, 10, 11, 12, 13, 14, 15);

        const size_t koff = (size_t)kt << 9;
        const bf16x16 f0 = *(const bf16x16*)(b0 + koff);
        const bf16x16 f1 = *(const bf16x16*)(b1 + koff);
        const bf16x16 f2 = *(const bf16x16*)(b2 + koff);
        const bf16x16 f3 = *(const bf16x16*)(b3 + koff);

        acc0 = __builtin_amdgcn_wmma_f32_16x16x32_bf16(false, a, false, f0, (short)0, acc0, false, false);
        acc1 = __builtin_amdgcn_wmma_f32_16x16x32_bf16(false, a, false, f1, (short)0, acc1, false, false);
        acc2 = __builtin_amdgcn_wmma_f32_16x16x32_bf16(false, a, false, f2, (short)0, acc2, false, false);
        acc3 = __builtin_amdgcn_wmma_f32_16x16x32_bf16(false, a, false, f3, (short)0, acc3, false, false);
    }

    // --- Store: C/D layout — lane half selects M 0-7 vs 8-15, VGPR v = M%8 ---
    #pragma unroll
    for (int t = 0; t < 4; ++t) {
        const f32x8 acc = (t == 0) ? acc0 : (t == 1) ? acc1 : (t == 2) ? acc2 : acc3;
        const int col = ((nt0 + t) << 4) + l15;
        const float bs = bias ? bias[col] : 0.0f;
        #pragma unroll
        for (int v = 0; v < 8; ++v) {
            const int row = m0 + (half << 3) + v;
            float val = acc[v] + bs;
            if (relu) val = fmaxf(val, 0.0f);
            const size_t idx = (size_t)row * N + col;
            if (Cf) Cf[idx] = val;
            if (Cb) Cb[idx] = f32_to_bf16(val);
        }
    }
}

// ---------------------------------------------------------------------------
// Block reductions (blockDim == 256)
// ---------------------------------------------------------------------------
__device__ __forceinline__ float block_reduce_max(float v, float* red) {
    int tid = threadIdx.x;
    red[tid] = v; __syncthreads();
    for (int s = 128; s > 0; s >>= 1) {
        if (tid < s) red[tid] = fmaxf(red[tid], red[tid + s]);
        __syncthreads();
    }
    float r = red[0]; __syncthreads();
    return r;
}
__device__ __forceinline__ float block_reduce_sum(float v, float* red) {
    int tid = threadIdx.x;
    red[tid] = v; __syncthreads();
    for (int s = 128; s > 0; s >>= 1) {
        if (tid < s) red[tid] += red[tid + s];
        __syncthreads();
    }
    float r = red[0]; __syncthreads();
    return r;
}

// ---------------------------------------------------------------------------
// One GRU time step for all 64 batch rows. Gate order z,r,h (Keras).
// z = softmax(xz+rz), r = softmax(xr+rr), hh = tanh(xh + r*rh),
// hn = z*h + (1-z)*hh.   One block per batch row (softmax over H=800).
// ---------------------------------------------------------------------------
__global__ void __launch_bounds__(256)
gru_step_kernel(const float* __restrict__ xg,      // [ROWS, 3H], row = b*T + t
                const float* __restrict__ rec,     // [B, 3H]  (h @ U + br)
                float* __restrict__ h,             // [B, H] fp32 state (in/out)
                unsigned short* __restrict__ hb,   // [B, H] bf16 state (out)
                unsigned short* __restrict__ yseq, // [ROWS, H] bf16 seq out
                int t, int T, int H)
{
    __shared__ float red[256];
    const int b = blockIdx.x, tid = threadIdx.x;
    const float* x  = xg  + ((size_t)b * T + t) * (3 * H);
    const float* rc = rec + (size_t)b * (3 * H);
    float* hp = h + (size_t)b * H;

    float sz[4], sr[4];
    int n = 0;
    float mz = -3.0e38f, mr = -3.0e38f;
    for (int i = tid; i < H; i += 256) {
        float vz = x[i] + rc[i];
        float vr = x[H + i] + rc[H + i];
        sz[n] = vz; sr[n] = vr; ++n;
        mz = fmaxf(mz, vz); mr = fmaxf(mr, vr);
    }
    mz = block_reduce_max(mz, red);
    mr = block_reduce_max(mr, red);

    float ez = 0.0f, er = 0.0f;
    for (int j = 0; j < n; ++j) {
        sz[j] = __expf(sz[j] - mz); ez += sz[j];
        sr[j] = __expf(sr[j] - mr); er += sr[j];
    }
    ez = block_reduce_sum(ez, red);
    er = block_reduce_sum(er, red);
    const float iz = 1.0f / ez, ir = 1.0f / er;

    n = 0;
    for (int i = tid; i < H; i += 256) {
        float z = sz[n] * iz, r = sr[n] * ir; ++n;
        float hh = tanhf(x[2 * H + i] + r * rc[2 * H + i]);
        float hn = z * hp[i] + (1.0f - z) * hh;
        hp[i] = hn;
        unsigned short bv = f32_to_bf16(hn);
        hb[(size_t)b * H + i] = bv;
        yseq[((size_t)b * T + t) * H + i] = bv;
    }
}

// ---------------------------------------------------------------------------
// Row softmax over N (in place), one block per row.
// ---------------------------------------------------------------------------
__global__ void __launch_bounds__(256)
softmax_rows(float* __restrict__ x, int N)
{
    __shared__ float red[256];
    float* row = x + (size_t)blockIdx.x * N;
    const int tid = threadIdx.x;
    float m = -3.0e38f;
    for (int i = tid; i < N; i += 256) m = fmaxf(m, row[i]);
    m = block_reduce_max(m, red);
    float s = 0.0f;
    for (int i = tid; i < N; i += 256) s += __expf(row[i] - m);
    s = block_reduce_sum(s, red);
    const float inv = 1.0f / s;
    for (int i = tid; i < N; i += 256) row[i] = __expf(row[i] - m) * inv;
}

// ---------------------------------------------------------------------------
// Embedding gather -> bf16 rows.  One block per output row.
// ---------------------------------------------------------------------------
__global__ void __launch_bounds__(256)
gather_rows_bf16(const int* __restrict__ idx, const float* __restrict__ emb,
                 unsigned short* __restrict__ out, int D)
{
    const int row = blockIdx.x;
    const float* src = emb + (size_t)idx[row] * D;
    unsigned short* dst = out + (size_t)row * D;
    for (int i = threadIdx.x; i < D; i += 256) dst[i] = f32_to_bf16(src[i]);
}

__global__ void __launch_bounds__(256)
zero_state(float* __restrict__ h, unsigned short* __restrict__ hb, int n)
{
    int i = blockIdx.x * 256 + threadIdx.x;
    if (i < n) { h[i] = 0.0f; hb[i] = 0; }
}

// ---------------------------------------------------------------------------
extern "C" void kernel_launch(void* const* d_in, const int* in_sizes, int n_in,
                              void* d_out, int out_size, void* d_ws, size_t ws_size,
                              hipStream_t stream)
{
    (void)in_sizes; (void)n_in; (void)out_size; (void)ws_size;

    const int*   enc    = (const int*)  d_in[0];
    const int*   dec    = (const int*)  d_in[1];
    const float* emb_fr = (const float*)d_in[2];
    const float* W_d1   = (const float*)d_in[3];
    const float* b_d1   = (const float*)d_in[4];
    const float* Wf1    = (const float*)d_in[5];
    const float* Uf1    = (const float*)d_in[6];
    const float* bf1    = (const float*)d_in[7];
    const float* Wf2    = (const float*)d_in[8];
    const float* Uf2    = (const float*)d_in[9];
    const float* bf2    = (const float*)d_in[10];
    const float* emb_en = (const float*)d_in[11];
    const float* We1    = (const float*)d_in[12];
    const float* Ue1    = (const float*)d_in[13];
    const float* be1    = (const float*)d_in[14];
    const float* We2    = (const float*)d_in[15];
    const float* Ue2    = (const float*)d_in[16];
    const float* be2    = (const float*)d_in[17];
    const float* W_out  = (const float*)d_in[18];
    const float* b_out  = (const float*)d_in[19];
    float* out = (float*)d_out;

    // ---- workspace carve (256B aligned) ----
    size_t off = 0;
    auto carve = [&](size_t bytes) -> void* {
        void* p = (char*)d_ws + off;
        off += (bytes + 255) & ~(size_t)255;
        return p;
    };
    unsigned short* wWd1  = (unsigned short*)carve((size_t)D_MODEL * D_MODEL * 2);
    unsigned short* wWf1  = (unsigned short*)carve((size_t)D_MODEL * H3 * 2);
    unsigned short* wUf1  = (unsigned short*)carve((size_t)D_MODEL * H3 * 2);
    unsigned short* wWf2  = (unsigned short*)carve((size_t)D_MODEL * H3 * 2);
    unsigned short* wUf2  = (unsigned short*)carve((size_t)D_MODEL * H3 * 2);
    unsigned short* wWe1  = (unsigned short*)carve((size_t)D_MODEL * H3 * 2);
    unsigned short* wUe1  = (unsigned short*)carve((size_t)D_MODEL * H3 * 2);
    unsigned short* wWe2  = (unsigned short*)carve((size_t)D_MODEL * H3 * 2);
    unsigned short* wUe2  = (unsigned short*)carve((size_t)D_MODEL * H3 * 2);
    unsigned short* wWout = (unsigned short*)carve((size_t)D_MODEL * VOUT * 2);
    unsigned short* seq   = (unsigned short*)carve((size_t)ROWS * D_MODEL * 2);  // fe/g1/g2/e1/e2 (bf16)
    unsigned short* embg  = (unsigned short*)carve((size_t)ROWS * D_MODEL * 2);  // gathered embeddings
    float*          xg    = (float*)carve((size_t)ROWS * H3 * 4);                // input-gate precompute
    float*          rec   = (float*)carve((size_t)BATCH * H3 * 4);               // h @ U + br
    float*          hf    = (float*)carve((size_t)BATCH * D_MODEL * 4);          // fp32 state
    unsigned short* hb    = (unsigned short*)carve((size_t)BATCH * D_MODEL * 2); // bf16 state

    // ---- pack weights fp32 -> bf16 fragment-major (stays L2-resident) ----
    struct { const float* s; unsigned short* d; int K, N; } pk[10] = {
        {W_d1,  wWd1,  D_MODEL, D_MODEL},
        {Wf1,   wWf1,  D_MODEL, H3}, {Uf1, wUf1, D_MODEL, H3},
        {Wf2,   wWf2,  D_MODEL, H3}, {Uf2, wUf2, D_MODEL, H3},
        {We1,   wWe1,  D_MODEL, H3}, {Ue1, wUe1, D_MODEL, H3},
        {We2,   wWe2,  D_MODEL, H3}, {Ue2, wUe2, D_MODEL, H3},
        {W_out, wWout, D_MODEL, VOUT},
    };
    for (int i = 0; i < 10; ++i) {
        long long n = (long long)pk[i].K * pk[i].N;
        long long g = (n + 255) / 256; if (g > 8192) g = 8192;
        pack_B_bf16<<<(int)g, 256, 0, stream>>>(pk[i].s, pk[i].d, pk[i].K, pk[i].N);
    }

    auto gemm = [&](const unsigned short* A, const unsigned short* Bp, const float* bias,
                    float* Cf, unsigned short* Cb, int M, int N, int K, int relu) {
        const int ntM = M / 16, ntN = N / 16;
        const int wavesN = (ntN + 3) / 4;
        const int waves = ntM * wavesN;
        const int blocks = (waves + 7) / 8;      // 8 wave32 per 256-thread block
        gemm_bf16_wmma<<<blocks, 256, 0, stream>>>(A, Bp, bias, Cf, Cb, M, N, K, relu);
    };

    auto run_layer = [&](const unsigned short* U, const float* brec) {
        for (int t = 0; t < TSEQ; ++t) {
            gemm(hb, U, brec, rec, nullptr, BATCH, H3, D_MODEL, 0);
            gru_step_kernel<<<BATCH, 256, 0, stream>>>(xg, rec, hf, hb, seq,
                                                       t, TSEQ, D_MODEL);
        }
    };

    // ---- encoder ----
    gather_rows_bf16<<<ROWS, 256, 0, stream>>>(enc, emb_fr, embg, D_MODEL);
    // fe = relu(embg @ W_d1 + b_d1)  (bf16 into seq)
    gemm(embg, wWd1, b_d1, nullptr, seq, ROWS, D_MODEL, D_MODEL, 1);
    // xg1 = fe @ Wf1 + bf1[0]
    gemm(seq, wWf1, bf1, xg, nullptr, ROWS, H3, D_MODEL, 0);
    zero_state<<<(BATCH * D_MODEL + 255) / 256, 256, 0, stream>>>(hf, hb, BATCH * D_MODEL);
    run_layer(wUf1, bf1 + H3);                      // -> g1 (seq), h1 (hf/hb)
    gemm(seq, wWf2, bf2, xg, nullptr, ROWS, H3, D_MODEL, 0);  // xg2 = g1 @ Wf2
    run_layer(wUf2, bf2 + H3);                      // -> h2 carries into decoder

    // ---- decoder ----
    gather_rows_bf16<<<ROWS, 256, 0, stream>>>(dec, emb_en, embg, D_MODEL);
    gemm(embg, wWe1, be1, xg, nullptr, ROWS, H3, D_MODEL, 0); // xg3 = ee @ We1
    run_layer(wUe1, be1 + H3);                      // -> e1 (seq), h3
    gemm(seq, wWe2, be2, xg, nullptr, ROWS, H3, D_MODEL, 0);  // xg4 = e1 @ We2
    run_layer(wUe2, be2 + H3);                      // -> e2 (seq), h4

    // ---- output projection + softmax (in place in d_out) ----
    gemm(seq, wWout, b_out, out, nullptr, ROWS, VOUT, D_MODEL, 0);
    softmax_rows<<<ROWS, 256, 0, stream>>>(out, VOUT);
}
